// NodeToEdgeAttr_71794673320040
// MI455X (gfx1250) — matrix operand order repaired and checked by reading, not compiled, not run on hardware
//
#include <hip/hip_runtime.h>

typedef __attribute__((ext_vector_type(16))) __bf16 v16bf;
typedef __attribute__((ext_vector_type(8)))  float  v8f;
typedef __attribute__((ext_vector_type(4)))  float  f4;

#define NN      50000
#define EE      800000
#define HID     128

// ---------- bf16 helpers ----------
// RNE conversion (prep kernel only; done once for the weights)
__device__ __forceinline__ unsigned short f2bfbits(float f) {
  unsigned u = __builtin_bit_cast(unsigned, f);
  u += 0x7FFFu + ((u >> 16) & 1u);
  return (unsigned short)(u >> 16);
}
__device__ __forceinline__ float bfbits2f(unsigned short s) {
  unsigned u = ((unsigned)s) << 16;
  return __builtin_bit_cast(float, u);
}

// Truncating split of a pair (a,b) -> packed hi / packed lo bf16 pairs.
// hi = trunc_bf16(v); lo = trunc_bf16(v - float(hi)).
// The hi truncation error is captured EXACTLY in lo (Sterbenz), so only lo's
// conversion error (<= 2^-16 relative) survives: fp32-class accuracy.
// Packing two high-halves is one v_perm_b32; float(hi) is one v_and_b32.
__device__ __forceinline__ void split_pair(float a, float b, unsigned& hi, unsigned& lo) {
  unsigned ua = __builtin_bit_cast(unsigned, a);
  unsigned ub = __builtin_bit_cast(unsigned, b);
  hi = __builtin_amdgcn_perm(ub, ua, 0x07060302u);  // [a.hi16, b.hi16]
  float la = a - __builtin_bit_cast(float, ua & 0xFFFF0000u);
  float lb = b - __builtin_bit_cast(float, ub & 0xFFFF0000u);
  unsigned ula = __builtin_bit_cast(unsigned, la);
  unsigned ulb = __builtin_bit_cast(unsigned, lb);
  lo = __builtin_amdgcn_perm(ulb, ula, 0x07060302u);
}

union V16U {
  v16bf v;
  unsigned u[8];
};

// v[16] in per-lane K order -> hi/lo v16bf operands (packed 32-bit writes only)
__device__ __forceinline__ void split16(const float* v, v16bf& hi, v16bf& lo) {
  V16U H, L;
#pragma unroll
  for (int p = 0; p < 8; ++p) split_pair(v[2 * p], v[2 * p + 1], H.u[p], L.u[p]);
  hi = H.v;
  lo = L.v;
}

// ------------------------------------------------------------------
// Prep: hi/lo bf16 copies of W_proj (256x256) and W_o (256x128), pre-swizzled
// into the WMMA B-operand lane layout:
//   tile (t = K/32, n = N/16), lane l, element j:
//     K = 32t + (l<16 ? j : 16+j),  N = 16n + (l&15)
// ------------------------------------------------------------------
__global__ __launch_bounds__(256) void prep_kernel(
    const float* __restrict__ Wproj, const float* __restrict__ Wo,
    unsigned short* __restrict__ g1hi, unsigned short* __restrict__ g1lo,
    unsigned short* __restrict__ g2hi, unsigned short* __restrict__ g2lo) {
  int tid = blockIdx.x * 256 + threadIdx.x;
  if (tid < 8 * 16 * 32 * 16) {  // W_proj: 8 K-tiles x 16 N-tiles
    int j = tid & 15, lane = (tid >> 4) & 31, n = (tid >> 9) & 15, t = tid >> 13;
    int K = 32 * t + ((lane < 16) ? j : 16 + j);
    int Nc = 16 * n + (lane & 15);
    float v = Wproj[K * 256 + Nc];
    unsigned short hb = f2bfbits(v);
    g1hi[tid] = hb;
    g1lo[tid] = f2bfbits(v - bfbits2f(hb));
  } else {  // W_o: 8 K-tiles x 8 N-tiles
    int id = tid - 8 * 16 * 32 * 16;
    if (id < 8 * 8 * 32 * 16) {
      int j = id & 15, lane = (id >> 4) & 31, n = (id >> 9) & 7, t = id >> 12;
      int K = 32 * t + ((lane < 16) ? j : 16 + j);
      int Nc = 16 * n + (lane & 15);
      float v = Wo[K * 128 + Nc];
      unsigned short hb = f2bfbits(v);
      g2hi[id] = hb;
      g2lo[id] = f2bfbits(v - bfbits2f(hb));
    }
  }
}

__device__ __forceinline__ v16bf lds_tile(const unsigned short* sB, int split, int t,
                                          int nt, int lane) {
  const unsigned short* p = sB + (((((split * 8 + t) * 4) + nt) * 32 + lane) * 16);
  return *(const v16bf*)p;
}

// ------------------------------------------------------------------
// GEMM1: qk = x @ W_proj + b_proj ; cols 0..127 -> q, 128..255 -> k
// 256 thr (8 waves); wave = 16 rows x 64 cols; blockIdx.y = N-quarter
// ------------------------------------------------------------------
__global__ __launch_bounds__(256) void gemm1_kernel(
    const float* __restrict__ x, const float* __restrict__ bproj,
    const unsigned short* __restrict__ g1hi, const unsigned short* __restrict__ g1lo,
    float* __restrict__ q, float* __restrict__ kvec) {
  __shared__ __attribute__((aligned(32))) unsigned short sB[32768];  // 64 KB
  const int tidb = threadIdx.x;
  const int by = blockIdx.y;

  {
    uint4* dst = (uint4*)sB;
    const uint4* hi4 = (const uint4*)g1hi;
    const uint4* lo4 = (const uint4*)g1lo;
    for (int idx = tidb; idx < 4096; idx += 256) {
      int chunk = idx >> 6, rest = idx & 63;
      int split = chunk >> 5, t = (chunk >> 2) & 7, nt = chunk & 3;
      const uint4* s = split ? lo4 : hi4;
      dst[idx] = s[(t * 16 + by * 4 + nt) * 64 + rest];
    }
  }
  __syncthreads();

  const int wave = tidb >> 5, lane = tidb & 31;
  const int lhalf = lane >> 4, lm = lane & 15;

  float bias[4];
#pragma unroll
  for (int nt = 0; nt < 4; ++nt) bias[nt] = bproj[by * 64 + nt * 16 + lm];

  const int nRowTiles = (NN + 127) / 128;  // 391
  for (int tile = blockIdx.x; tile < nRowTiles; tile += gridDim.x) {
    const int rowBase = tile * 128 + wave * 16;
    if (rowBase >= NN) continue;  // NN % 16 == 0: slab fully valid or fully out
    const float* xr = x + (long)(rowBase + lm) * 256;

    v8f acc[4];
#pragma unroll
    for (int nt = 0; nt < 4; ++nt)
#pragma unroll
      for (int r = 0; r < 8; ++r) acc[nt][r] = bias[nt];

    for (int t = 0; t < 8; ++t) {  // runtime loop: bounded live range
      const int k0 = 32 * t + (lhalf ? 16 : 0);
      f4 a0 = *(const f4*)(xr + k0 + 0);
      f4 a1 = *(const f4*)(xr + k0 + 4);
      f4 a2 = *(const f4*)(xr + k0 + 8);
      f4 a3 = *(const f4*)(xr + k0 + 12);
      float v[16] = {a0.x, a0.y, a0.z, a0.w, a1.x, a1.y, a1.z, a1.w,
                     a2.x, a2.y, a2.z, a2.w, a3.x, a3.y, a3.z, a3.w};
      v16bf ahi, alo;
      split16(v, ahi, alo);
#pragma unroll
      for (int nt = 0; nt < 4; ++nt) {
        v16bf bhi = lds_tile(sB, 0, t, nt, lane);
        acc[nt] = __builtin_amdgcn_wmma_f32_16x16x32_bf16(false, ahi, false, bhi,
                                                          (short)0, acc[nt], false, false);
        acc[nt] = __builtin_amdgcn_wmma_f32_16x16x32_bf16(false, alo, false, bhi,
                                                          (short)0, acc[nt], false, false);
        v16bf blo = lds_tile(sB, 1, t, nt, lane);
        acc[nt] = __builtin_amdgcn_wmma_f32_16x16x32_bf16(false, ahi, false, blo,
                                                          (short)0, acc[nt], false, false);
      }
    }

#pragma unroll
    for (int nt = 0; nt < 4; ++nt) {
      const int col = by * 64 + nt * 16 + lm;
      float* ob = (col < HID) ? (q + col) : (kvec + (col - HID));
#pragma unroll
      for (int r = 0; r < 8; ++r) {
        int row = rowBase + (lhalf ? r + 8 : r);
        ob[(long)row * HID] = acc[nt][r];
      }
    }
  }
}

// ------------------------------------------------------------------
// GEMM2: out = concat(q[src]*k[dst], q[src]-k[dst]) @ W_o + b_o
// wave = 16 edges x 64 cols; blockIdx.y = N-half; grid-stride over edge tiles
// ------------------------------------------------------------------
__global__ __launch_bounds__(256) void gemm2_kernel(
    const float* __restrict__ q, const float* __restrict__ kvec,
    const int* __restrict__ ei, const float* __restrict__ bo,
    const unsigned short* __restrict__ g2hi, const unsigned short* __restrict__ g2lo,
    float* __restrict__ out) {
  __shared__ __attribute__((aligned(32))) unsigned short sB[32768];  // 64 KB
  const int tidb = threadIdx.x;
  const int half = blockIdx.y;

  {
    uint4* dst = (uint4*)sB;
    const uint4* hi4 = (const uint4*)g2hi;
    const uint4* lo4 = (const uint4*)g2lo;
    for (int idx = tidb; idx < 4096; idx += 256) {
      int chunk = idx >> 6, rest = idx & 63;
      int split = chunk >> 5, t = (chunk >> 2) & 7, nt = chunk & 3;
      const uint4* s = split ? lo4 : hi4;
      dst[idx] = s[(t * 8 + half * 4 + nt) * 64 + rest];
    }
  }
  __syncthreads();

  const int wave = tidb >> 5, lane = tidb & 31;
  const int lhalf = lane >> 4, lm = lane & 15;

  float bias[4];
#pragma unroll
  for (int nt = 0; nt < 4; ++nt) bias[nt] = bo[half * 64 + nt * 16 + lm];

  const int nTiles = EE / 128;  // 6250 exactly
  for (int tile = blockIdx.x; tile < nTiles; tile += gridDim.x) {
    const int e0 = tile * 128 + wave * 16;
    const int e = e0 + lm;
    const int srcN = ei[e];
    const int dstN = ei[EE + e];
    const float* qr = q + (long)srcN * HID;
    const float* kr = kvec + (long)dstN * HID;

    v8f acc[4];
#pragma unroll
    for (int nt = 0; nt < 4; ++nt)
#pragma unroll
      for (int r = 0; r < 8; ++r) acc[nt][r] = bias[nt];

    for (int tf = 0; tf < 4; ++tf) {  // runtime loop: bounded live range
      const int f0 = 32 * tf + (lhalf ? 16 : 0);
      f4 q0 = *(const f4*)(qr + f0 + 0);
      f4 q1 = *(const f4*)(qr + f0 + 4);
      f4 q2 = *(const f4*)(qr + f0 + 8);
      f4 q3 = *(const f4*)(qr + f0 + 12);
      f4 k0 = *(const f4*)(kr + f0 + 0);
      f4 k1 = *(const f4*)(kr + f0 + 4);
      f4 k2 = *(const f4*)(kr + f0 + 8);
      f4 k3 = *(const f4*)(kr + f0 + 12);
      float qv[16] = {q0.x, q0.y, q0.z, q0.w, q1.x, q1.y, q1.z, q1.w,
                      q2.x, q2.y, q2.z, q2.w, q3.x, q3.y, q3.z, q3.w};
      float kv[16] = {k0.x, k0.y, k0.z, k0.w, k1.x, k1.y, k1.z, k1.w,
                      k2.x, k2.y, k2.z, k2.w, k3.x, k3.y, k3.z, k3.w};
      float pv[16], dv[16];
#pragma unroll
      for (int j = 0; j < 16; ++j) {
        pv[j] = qv[j] * kv[j];
        dv[j] = qv[j] - kv[j];
      }
      v16bf aph, apl, adh, adl;
      split16(pv, aph, apl);
      split16(dv, adh, adl);
#pragma unroll
      for (int nt = 0; nt < 4; ++nt) {
        v16bf bph = lds_tile(sB, 0, tf, nt, lane);  // product half: K-tile tf
        acc[nt] = __builtin_amdgcn_wmma_f32_16x16x32_bf16(false, aph, false, bph,
                                                          (short)0, acc[nt], false, false);
        acc[nt] = __builtin_amdgcn_wmma_f32_16x16x32_bf16(false, apl, false, bph,
                                                          (short)0, acc[nt], false, false);
        v16bf bpl = lds_tile(sB, 1, tf, nt, lane);
        acc[nt] = __builtin_amdgcn_wmma_f32_16x16x32_bf16(false, aph, false, bpl,
                                                          (short)0, acc[nt], false, false);
        v16bf bdh = lds_tile(sB, 0, tf + 4, nt, lane);  // diff half: K-tile tf+4
        acc[nt] = __builtin_amdgcn_wmma_f32_16x16x32_bf16(false, adh, false, bdh,
                                                          (short)0, acc[nt], false, false);
        acc[nt] = __builtin_amdgcn_wmma_f32_16x16x32_bf16(false, adl, false, bdh,
                                                          (short)0, acc[nt], false, false);
        v16bf bdl = lds_tile(sB, 1, tf + 4, nt, lane);
        acc[nt] = __builtin_amdgcn_wmma_f32_16x16x32_bf16(false, adh, false, bdl,
                                                          (short)0, acc[nt], false, false);
      }
    }

#pragma unroll
    for (int nt = 0; nt < 4; ++nt) {
      const int col = half * 64 + nt * 16 + lm;
#pragma unroll
      for (int r = 0; r < 8; ++r) {
        int er = e0 + (lhalf ? r + 8 : r);
        out[(long)er * HID + col] = acc[nt][r];
      }
    }
  }
}

// ------------------------------------------------------------------
extern "C" void kernel_launch(void* const* d_in, const int* in_sizes, int n_in,
                              void* d_out, int out_size, void* d_ws, size_t ws_size,
                              hipStream_t stream) {
  const float* x     = (const float*)d_in[0];
  const int*   ei    = (const int*)d_in[1];
  const float* Wproj = (const float*)d_in[2];
  const float* bproj = (const float*)d_in[3];
  const float* Wo    = (const float*)d_in[4];
  const float* bo    = (const float*)d_in[5];
  float* out = (float*)d_out;

  char* ws = (char*)d_ws;
  float* q  = (float*)(ws);                                 // 25,600,000 B
  float* kv = (float*)(ws + 25600000);                      // 25,600,000 B
  unsigned short* g1hi = (unsigned short*)(ws + 51200000);  // 128 KB
  unsigned short* g1lo = g1hi + 65536;                      // 128 KB
  unsigned short* g2hi = g1lo + 65536;                      // 64 KB
  unsigned short* g2lo = g2hi + 32768;                      // 64 KB

  prep_kernel<<<384, 256, 0, stream>>>(Wproj, Wo, g1hi, g1lo, g2hi, g2lo);
  gemm1_kernel<<<dim3(391, 4), 256, 0, stream>>>(x, bproj, g1hi, g1lo, q, kv);
  gemm2_kernel<<<dim3(640, 2), 256, 0, stream>>>(q, kv, ei, bo, g2hi, g2lo, out);
}
